// EncoderLayer_12979391168612
// MI455X (gfx1250) — matrix operand order repaired and checked by reading, not compiled, not run on hardware
//
#include <hip/hip_runtime.h>

typedef __attribute__((ext_vector_type(16))) __bf16 v16bf;
typedef __attribute__((ext_vector_type(8)))  float  v8f;

#define BDIM 2
#define CDIM 2048
#define MDIM 1024
#define HDIM 16
#define KHD  64      // head dim (K)
#define VHD  64      // head value dim
#define FDIM 4096
#define HV   (HDIM*VHD)   // 1024
#define LN_EPS 1e-5f

enum { F_RELU = 1, F_BIAS = 2, F_RES = 4, F_OUTF = 8, F_OUTB = 16 };

static __device__ inline v8f zero8() {
  v8f z = {0.f,0.f,0.f,0.f,0.f,0.f,0.f,0.f};
  return z;
}

static __device__ inline v8f wmma_bf16(v16bf a, v16bf b, v8f c) {
  // D = A(16x32 bf16) * B(32x16 bf16) + C(16x16 f32)
  return __builtin_amdgcn_wmma_f32_16x16x32_bf16(
      /*neg_a=*/false, a, /*neg_b=*/false, b,
      /*c_mod=*/(short)0, c, /*reuse_a=*/false, /*reuse_b=*/false);
}

// Async global->LDS copy (16B per lane), tracked by ASYNCcnt (CDNA5).
// VDST VGPR = LDS byte address (low 32 bits of generic pointer to __shared__).
static __device__ inline void async_load_b128(void* lds_dst, const void* gsrc) {
  unsigned lds = (unsigned)(size_t)lds_dst;
  unsigned long long ga = (unsigned long long)(size_t)gsrc;
  asm volatile("global_load_async_to_lds_b128 %0, %1, off"
               :: "v"(lds), "v"(ga) : "memory");
}
static __device__ inline void wait_async0() {
  asm volatile("s_wait_asynccnt 0x0" ::: "memory");
}

// A fragment (16x32, row-major source with 'stride' elements per row).
// lane r = lane&15 -> row; lane half selects K sub-range:
// VGPR i holds pair at k = (i>>2)*16 + half*8 + (i&3)*2
static __device__ inline v16bf load_frag_a(const __bf16* p, int stride,
                                           int row0, int kbase, int lane) {
  int r = lane & 15, hf = lane >> 4;
  const __bf16* base = p + (size_t)(row0 + r) * stride + kbase + (hf << 3);
  v16bf a;
#pragma unroll
  for (int i = 0; i < 8; ++i) {
    int k = ((i >> 2) << 4) + ((i & 3) << 1);
    a[2*i]   = base[k];
    a[2*i+1] = base[k+1];
  }
  return a;
}

// B fragment (32x16) from K-contiguous storage p[col][k] ('stride' elems/col).
// lane col = lane&15; element j holds k = (j>>3)*16 + half*8 + (j&7).
static __device__ inline v16bf load_frag_b(const __bf16* p, int stride,
                                           int col0, int kbase, int lane) {
  int c = lane & 15, hf = lane >> 4;
  const __bf16* base = p + (size_t)(col0 + c) * stride + kbase + (hf << 3);
  v16bf b;
#pragma unroll
  for (int j = 0; j < 16; ++j) b[j] = base[((j >> 3) << 4) + (j & 7)];
  return b;
}

// ---------------------------------------------------------------------------
// f32 -> bf16 conversion (grid-stride)
// ---------------------------------------------------------------------------
__global__ void cvt_f32_bf16(const float* __restrict__ in,
                             __bf16* __restrict__ out, long n) {
  long i = (long)blockIdx.x * blockDim.x + threadIdx.x;
  long stride = (long)gridDim.x * blockDim.x;
  for (; i < n; i += stride) out[i] = (__bf16)in[i];
}

// ---------------------------------------------------------------------------
// Generic bf16 WMMA GEMM, compile-time fused epilogue (FLAGS template arg).
// Block tile 256x64, K-step 32, 8 waves; wave = 32 rows x 64 cols
// (2 A-fragments x 4 B-fragments = 8 WMMAs per K-step).
// A tile staged with async global->LDS (ASYNCcnt); B tile staged transposed.
// Batched via blockIdx.z:  A += (z/aZdiv)*aZstride, B += (z%bZmod)*bZstride,
// C += z*cZstride (also applied to residual).
// ---------------------------------------------------------------------------
template <int FLAGS>
__global__ __launch_bounds__(256) void gemm_bf16_wmma(
    const __bf16* __restrict__ A, const __bf16* __restrict__ Bm,
    float* __restrict__ Cf, __bf16* __restrict__ Cb,
    const float* __restrict__ bias, const float* __restrict__ Res,
    int Kd, int lda, int ldb, int ldc,
    long aZstride, int aZdiv, long bZstride, int bZmod,
    long cZstride)
{
  __shared__ __bf16 sA[256 * 32];   // A tile, row-major [256][32]
  __shared__ __bf16 sBt[64 * 32];   // B tile transposed [col][k]

  int z = blockIdx.z;
  A  += (long)(z / aZdiv) * aZstride;
  Bm += (long)(z % bZmod) * bZstride;
  long coff = (long)z * cZstride;

  int tid = threadIdx.x, w = tid >> 5, lane = tid & 31;
  int row0 = blockIdx.x * 256, col0 = blockIdx.y * 64;

  // A staging map: fully coalesced (adjacent threads -> adjacent 16B of a row)
  int ar = tid >> 2, ac = (tid & 3) << 3;        // 64 rows per pass, 4 passes
  // B staging map: threads 0..127, each covers 2 k-rows x 8 cols
  int bk = (tid >> 3) << 1, bc = (tid & 7) << 3; // bk in {0..30}, valid tid<128

  v8f acc[2][4];
#pragma unroll
  for (int mt = 0; mt < 2; ++mt)
#pragma unroll
    for (int nt = 0; nt < 4; ++nt) acc[mt][nt] = zero8();

  for (int k0 = 0; k0 < Kd; k0 += 32) {
    __syncthreads();
    // ---- stage A: 256x32 bf16 via async global->LDS (4 x 16B per thread) ----
#pragma unroll
    for (int j = 0; j < 4; ++j) {
      int r = ar + j * 64;
      async_load_b128(&sA[r * 32 + ac],
                      A + (long)(row0 + r) * lda + k0 + ac);
    }
    // ---- stage B transposed (pairs of k packed into u32 stores) ----
    if (tid < 128) {
      union { uint4 u; __bf16 h[8]; } t0, t1;
      t0.u = *(const uint4*)(Bm + (long)(k0 + bk)     * ldb + col0 + bc);
      t1.u = *(const uint4*)(Bm + (long)(k0 + bk + 1) * ldb + col0 + bc);
#pragma unroll
      for (int j = 0; j < 8; ++j) {
        union { unsigned u; __bf16 h[2]; } p;
        p.h[0] = t0.h[j];
        p.h[1] = t1.h[j];
        *(unsigned*)&sBt[(bc + j) * 32 + bk] = p.u;
      }
    }
    // ---- prefetch next K tile into cache (global_prefetch_b8) ----
    if (k0 + 32 < Kd) {
      __builtin_prefetch(A + (long)(row0 + ar) * lda + k0 + 32 + ac, 0, 3);
      if (tid < 128)
        __builtin_prefetch(Bm + (long)(k0 + 32 + bk) * ldb + col0 + bc, 0, 3);
    }
    wait_async0();
    __syncthreads();

    v16bf a0 = load_frag_a(sA, 32, w * 32,      0, lane);
    v16bf a1 = load_frag_a(sA, 32, w * 32 + 16, 0, lane);
#pragma unroll
    for (int nt = 0; nt < 4; ++nt) {
      v16bf b = load_frag_b(sBt, 32, nt * 16, 0, lane);
      acc[0][nt] = wmma_bf16(a0, b, acc[0][nt]);
      acc[1][nt] = wmma_bf16(a1, b, acc[1][nt]);
    }
  }

  int hf = lane >> 4, cl = lane & 15;
#pragma unroll
  for (int mt = 0; mt < 2; ++mt) {
#pragma unroll
    for (int nt = 0; nt < 4; ++nt) {
#pragma unroll
      for (int i = 0; i < 8; ++i) {
        int r = row0 + w * 32 + mt * 16 + i + 8 * hf;
        int c = col0 + nt * 16 + cl;
        float v = acc[mt][nt][i];
        if (FLAGS & F_BIAS) v += bias[c];
        if (FLAGS & F_RELU) v = fmaxf(v, 0.f);
        long idx = coff + (long)r * ldc + c;
        if (FLAGS & F_RES)  v += Res[idx];
        if (FLAGS & F_OUTF) Cf[idx] = v;
        if (FLAGS & F_OUTB) Cb[idx] = (__bf16)v;
      }
    }
  }
}

// ---------------------------------------------------------------------------
// Flash-style attention. One block per (128 queries, b*H+h). 8 waves of 16
// queries. S computed in [key, query] orientation so per-query softmax stats
// need only 8 in-lane regs + one shfl_xor(16). Soft causal mask (-100 pre-
// scale == -12.5 post-scale). K tile staged async; V staged transposed.
// Output written as preb[b][query][h*64+v] (bf16) for the out-proj GEMM.
// ---------------------------------------------------------------------------
__global__ __launch_bounds__(256) void attention_kernel(
    const __bf16* __restrict__ Qb, const __bf16* __restrict__ Kb,
    const __bf16* __restrict__ Vb, __bf16* __restrict__ preb)
{
  __shared__ __bf16 sK[32 * 64];        // 32 keys x 64 k, row-major
  __shared__ __bf16 sVt[64 * 32];       // V transposed: [v][key]
  __shared__ __bf16 sP[8][16 * 32];     // per-wave P: [query][key]

  int z = blockIdx.y;                   // b*H + h
  int b = z / HDIM, hh = z % HDIM;
  long zoff = (long)z * CDIM * KHD;

  int tid = threadIdx.x, w = tid >> 5, lane = tid & 31;
  int hf = lane >> 4, cl = lane & 15;
  int q0 = blockIdx.x * 128 + w * 16;   // wave's first query
  int dg = q0 + cl;                     // this lane's query index

  // Q fragments resident for the whole key loop (B operand, cols = queries)
  v16bf bq0 = load_frag_b(Qb + zoff, KHD, q0, 0,  lane);
  v16bf bq1 = load_frag_b(Qb + zoff, KHD, q0, 32, lane);

  v8f accv[4] = {zero8(), zero8(), zero8(), zero8()}; // preT[v][query], v=64
  float m = -1e30f, l = 0.f;

  int sr = tid >> 3, sc = (tid & 7) << 3;   // K/V staging map

  for (int kb = 0; kb < CDIM / 32; ++kb) {
    __syncthreads();
    { // stage K tile async (16B/thread); V tile transposed (8 bf16/thread)
      async_load_b128(&sK[sr * 64 + sc],
                      Kb + zoff + (long)(kb * 32 + sr) * KHD + sc);
      union { uint4 u; __bf16 h[8]; } t;
      t.u = *(const uint4*)(Vb + zoff + (long)(kb * 32 + sr) * VHD + sc);
#pragma unroll
      for (int j = 0; j < 8; ++j) sVt[(sc + j) * 32 + sr] = t.h[j];
      if (kb + 1 < CDIM / 32) { // prefetch next key block
        __builtin_prefetch(Kb + zoff + (long)((kb + 1) * 32 + sr) * KHD + sc, 0, 3);
        __builtin_prefetch(Vb + zoff + (long)((kb + 1) * 32 + sr) * VHD + sc, 0, 3);
      }
    }
    wait_async0();
    __syncthreads();

    // S[key, query] for two 16-key sub-tiles, K-dim = head dim 64 (2 steps)
    v8f s0 = wmma_bf16(load_frag_a(sK, 64, 0, 0, lane), bq0, zero8());
    s0     = wmma_bf16(load_frag_a(sK, 64, 0, 32, lane), bq1, s0);
    v8f s1 = wmma_bf16(load_frag_a(sK, 64, 16, 0, lane), bq0, zero8());
    s1     = wmma_bf16(load_frag_a(sK, 64, 16, 32, lane), bq1, s1);

    // masked + scaled logits; online max
    float t0[8], t1[8];
    float lm = -1e30f;
    int cbase = kb * 32 + 8 * hf;
#pragma unroll
    for (int i = 0; i < 8; ++i) {
      int cg0 = cbase + i, cg1 = cbase + 16 + i;
      t0[i] = s0[i] * 0.125f + (cg0 > dg ? -12.5f : 0.f);
      t1[i] = s1[i] * 0.125f + (cg1 > dg ? -12.5f : 0.f);
      lm = fmaxf(lm, fmaxf(t0[i], t1[i]));
    }
    lm = fmaxf(lm, __shfl_xor(lm, 16, 32));
    float mn = fmaxf(m, lm);
    float corr = __expf(m - mn);

    float ps = 0.f;
#pragma unroll
    for (int i = 0; i < 8; ++i) {
      float p0 = __expf(t0[i] - mn);
      float p1 = __expf(t1[i] - mn);
      ps += p0 + p1;
      sP[w][cl * 32 + (8 * hf + i)]        = (__bf16)p0;
      sP[w][cl * 32 + (16 + 8 * hf + i)]   = (__bf16)p1;
    }
    ps += __shfl_xor(ps, 16, 32);
    l = l * corr + ps;
    m = mn;

    // preT[v][d] += V^T(16v x 32c) @ P(32c x 16d)
    v16bf bp = load_frag_b(sP[w], 32, 0, 0, lane);
#pragma unroll
    for (int vt = 0; vt < 4; ++vt) {
#pragma unroll
      for (int i = 0; i < 8; ++i) accv[vt][i] *= corr;
      v16bf av = load_frag_a(sVt, 32, vt * 16, 0, lane);
      accv[vt] = wmma_bf16(av, bp, accv[vt]);
    }
  }

  float rl = 1.f / l;
  long obase = ((long)b * CDIM + dg) * HV + (long)hh * VHD;
#pragma unroll
  for (int vt = 0; vt < 4; ++vt)
#pragma unroll
    for (int i = 0; i < 8; ++i)
      preb[obase + vt * 16 + 8 * hf + i] = (__bf16)(accv[vt][i] * rl);
}

// ---------------------------------------------------------------------------
// LayerNorm over axis=1 (sequence axis C), params broadcast on M.
// grid (M/64, B), block (64, 4): threadIdx.x -> feature m (coalesced),
// threadIdx.y -> 4-way partial over c, reduced via LDS.
// ---------------------------------------------------------------------------
__global__ __launch_bounds__(256) void layernorm_c(
    const float* __restrict__ X, const float* __restrict__ scale,
    const float* __restrict__ offset, float* __restrict__ Of,
    __bf16* __restrict__ Ob)
{
  int mcol = blockIdx.x * 64 + threadIdx.x;
  int b = blockIdx.y;
  int ty = threadIdx.y;
  const float* col = X + (long)b * CDIM * MDIM + mcol;

  float s = 0.f, ss = 0.f;
  for (int c = ty; c < CDIM; c += 4) {
    float v = col[(long)c * MDIM];
    s += v;
    ss += v * v;
  }
  __shared__ float r1[4][64], r2[4][64];
  r1[ty][threadIdx.x] = s;
  r2[ty][threadIdx.x] = ss;
  __syncthreads();
  float S  = r1[0][threadIdx.x] + r1[1][threadIdx.x] +
             r1[2][threadIdx.x] + r1[3][threadIdx.x];
  float SS = r2[0][threadIdx.x] + r2[1][threadIdx.x] +
             r2[2][threadIdx.x] + r2[3][threadIdx.x];
  float mean = S * (1.f / CDIM);
  float var  = SS * (1.f / CDIM) - mean * mean;
  float rstd = rsqrtf(var + LN_EPS);
  float sc = scale[mcol] * rstd;
  float of = offset[mcol] - mean * sc;

  for (int c = ty; c < CDIM; c += 4) {
    long idx = (long)b * CDIM * MDIM + (long)c * MDIM + mcol;
    float v = X[idx] * sc + of;
    if (Of) Of[idx] = v;
    if (Ob) Ob[idx] = (__bf16)v;
  }
}

// ---------------------------------------------------------------------------
// Host orchestration
// ---------------------------------------------------------------------------
extern "C" void kernel_launch(void* const* d_in, const int* in_sizes, int n_in,
                              void* d_out, int out_size, void* d_ws, size_t ws_size,
                              hipStream_t stream) {
  (void)in_sizes; (void)n_in; (void)out_size; (void)ws_size;
  const float* x    = (const float*)d_in[0];
  const float* wq   = (const float*)d_in[1];
  const float* wk   = (const float*)d_in[2];
  const float* wv   = (const float*)d_in[3];
  const float* wo   = (const float*)d_in[4];
  const float* ln1s = (const float*)d_in[5];
  const float* ln1o = (const float*)d_in[6];
  const float* w1   = (const float*)d_in[7];
  const float* b1   = (const float*)d_in[8];
  const float* w2   = (const float*)d_in[9];
  const float* b2   = (const float*)d_in[10];
  const float* ln2s = (const float*)d_in[11];
  const float* ln2o = (const float*)d_in[12];
  float* out = (float*)d_out;

  char* ws = (char*)d_ws;
  size_t off = 0;
  auto alloc = [&](size_t bytes) {
    size_t o = off;
    off += (bytes + 255) & ~(size_t)255;
    return ws + o;
  };

  const long nX  = (long)BDIM * CDIM * MDIM;         // 4.19M
  const long nWq = (long)HDIM * MDIM * KHD;          // 1.05M
  const long nWo = (long)HDIM * VHD * MDIM;          // 1.05M
  const long nW1 = (long)MDIM * FDIM;                // 4.19M
  const long nQ  = (long)BDIM * HDIM * CDIM * KHD;   // 4.19M
  const long nPre= (long)BDIM * CDIM * HV;           // 4.19M
  const long nS  = (long)BDIM * CDIM * FDIM;         // 16.8M

  __bf16* xb   = (__bf16*)alloc(nX  * 2);
  __bf16* wqb  = (__bf16*)alloc(nWq * 2);
  __bf16* wkb  = (__bf16*)alloc(nWq * 2);
  __bf16* wvb  = (__bf16*)alloc(nWq * 2);
  __bf16* wob  = (__bf16*)alloc(nWo * 2);  // flattened [HV, M]
  __bf16* w1b  = (__bf16*)alloc(nW1 * 2);
  __bf16* w2b  = (__bf16*)alloc(nW1 * 2);
  __bf16* Qb   = (__bf16*)alloc(nQ  * 2);  // [B,H,C,64]
  __bf16* Kb   = (__bf16*)alloc(nQ  * 2);
  __bf16* Vb   = (__bf16*)alloc(nQ  * 2);
  __bf16* preb = (__bf16*)alloc(nPre* 2);  // [B,C,HV]
  float*  y1    = (float*)alloc(nX * 4);   // x + att_out
  float*  normf = (float*)alloc(nX * 4);   // ln1 output (f32, residual source)
  __bf16* normb = (__bf16*)alloc(nX * 2);  // ln1 output (bf16, GEMM source)
  __bf16* sb   = (__bf16*)alloc(nS  * 2);  // relu(norm@w1+b1)
  float*  y2    = (float*)alloc(nX * 4);   // norm + ffn

  // 1) precision conversion
  cvt_f32_bf16<<<dim3(512), dim3(256), 0, stream>>>(x,  xb,  nX);
  cvt_f32_bf16<<<dim3(256), dim3(256), 0, stream>>>(wq, wqb, nWq);
  cvt_f32_bf16<<<dim3(256), dim3(256), 0, stream>>>(wk, wkb, nWq);
  cvt_f32_bf16<<<dim3(256), dim3(256), 0, stream>>>(wv, wvb, nWq);
  cvt_f32_bf16<<<dim3(256), dim3(256), 0, stream>>>(wo, wob, nWo);
  cvt_f32_bf16<<<dim3(512), dim3(256), 0, stream>>>(w1, w1b, nW1);
  cvt_f32_bf16<<<dim3(512), dim3(256), 0, stream>>>(w2, w2b, nW1);

  // 2) QKV projections: per (b,h): [C,1024] @ [1024,64] -> [C,64] bf16
  const long aZs = (long)CDIM * MDIM;   // per-b x stride
  const long bZs = (long)MDIM * KHD;    // per-h weight stride
  const long cZs = (long)CDIM * KHD;    // per-(b,h) output stride
  dim3 gQKV(CDIM / 256, 1, BDIM * HDIM);
  gemm_bf16_wmma<F_OUTB><<<gQKV, 256, 0, stream>>>(
      xb, wqb, nullptr, Qb, nullptr, nullptr,
      MDIM, MDIM, KHD, KHD, aZs, HDIM, bZs, HDIM, cZs);
  gemm_bf16_wmma<F_OUTB><<<gQKV, 256, 0, stream>>>(
      xb, wkb, nullptr, Kb, nullptr, nullptr,
      MDIM, MDIM, KHD, KHD, aZs, HDIM, bZs, HDIM, cZs);
  gemm_bf16_wmma<F_OUTB><<<gQKV, 256, 0, stream>>>(
      xb, wvb, nullptr, Vb, nullptr, nullptr,
      MDIM, MDIM, KHD, KHD, aZs, HDIM, bZs, HDIM, cZs);

  // 3) attention -> preb [B,C,HV]
  attention_kernel<<<dim3(CDIM / 128, BDIM * HDIM), 256, 0, stream>>>(
      Qb, Kb, Vb, preb);

  // 4) output projection + residual: y1 = x + preb @ wo_flat ([4096,1024]@[1024,1024])
  gemm_bf16_wmma<F_RES | F_OUTF><<<dim3(BDIM * CDIM / 256, MDIM / 64, 1), 256, 0, stream>>>(
      preb, wob, y1, nullptr, nullptr, x,
      HV, HV, MDIM, MDIM, 0, 1, 0, 1, 0);

  // 5) layernorm over sequence axis
  layernorm_c<<<dim3(MDIM / 64, BDIM), dim3(64, 4), 0, stream>>>(
      y1, ln1s, ln1o, normf, normb);

  // 6) FFN up: sb = relu(norm @ w1 + b1)  ([4096,1024]@[1024,4096])
  gemm_bf16_wmma<F_BIAS | F_RELU | F_OUTB><<<dim3(BDIM * CDIM / 256, FDIM / 64, 1), 256, 0, stream>>>(
      normb, w1b, nullptr, sb, b1, nullptr,
      MDIM, MDIM, FDIM, FDIM, 0, 1, 0, 1, 0);

  // 7) FFN down + residual: y2 = norm + sb @ w2 + b2  ([4096,4096]@[4096,1024])
  gemm_bf16_wmma<F_BIAS | F_RES | F_OUTF><<<dim3(BDIM * CDIM / 256, MDIM / 64, 1), 256, 0, stream>>>(
      sb, w2b, y2, nullptr, b2, normf,
      FDIM, FDIM, MDIM, MDIM, 0, 1, 0, 1, 0);

  // 8) final layernorm -> output (f32)
  layernorm_c<<<dim3(MDIM / 64, BDIM), dim3(64, 4), 0, stream>>>(
      y2, ln2s, ln2o, out, nullptr);
}
